// ConnectionsSMPV2_83760452206683
// MI455X (gfx1250) — matrix operand order repaired and checked by reading, not compiled, not run on hardware
//
#include <hip/hip_runtime.h>
#include <hip/hip_bf16.h>

#define NOBJ   32
#define NSEG   33
#define FDIM   64
#define HWPIX  65536
#define BSZ    8
#define CHUNKS 32
#define LSTRIDE 65   // 33 x 65 padded LDS table -> bank = (id+f)%64

typedef __attribute__((ext_vector_type(16))) _Float16 v16h;
typedef __attribute__((ext_vector_type(8)))  float    v8f;

__global__ void init_vec_kernel(float* __restrict__ vec) {
    int idx = blockIdx.x * blockDim.x + threadIdx.x;
    if (idx < BSZ * NOBJ * FDIM) vec[idx] = -__builtin_inff();
}

// Per-block segment-max over a pixel chunk; LDS float atomics, then global
// float atomic-max flush into the vectors region of d_out.
__global__ __launch_bounds__(256) void segmax_kernel(const float* __restrict__ enc,
                                                     const int*   __restrict__ ids,
                                                     float*       __restrict__ vec) {
    __shared__ float lmax[NSEG * LSTRIDE];
    const int t = threadIdx.x;
    for (int i = t; i < NSEG * LSTRIDE; i += 256) lmax[i] = -__builtin_inff();
    __syncthreads();

    const int b     = blockIdx.x / CHUNKS;
    const int chunk = blockIdx.x % CHUNKS;
    const int PPB   = HWPIX / CHUNKS;           // 2048 pixels per block
    const int pbase = chunk * PPB;

    const float* encb = enc + (size_t)b * FDIM * HWPIX;
    const int*   idsb = ids + (size_t)b * HWPIX;

    for (int p = pbase + t; p < pbase + PPB; p += 256) {
        const int id = idsb[p];
        float* row = &lmax[id * LSTRIDE];
        #pragma unroll
        for (int f = 0; f < FDIM; ++f) {
            float v = encb[(size_t)f * HWPIX + p];   // lanes -> consecutive p: coalesced
            atomicMax(&row[f], v);
        }
    }
    __syncthreads();

    for (int i = t; i < NSEG * FDIM; i += 256) {
        const int id = i >> 6;
        const int f  = i & 63;
        if (id >= 1) {
            atomicMax(&vec[((size_t)b * NOBJ + (id - 1)) * FDIM + f],
                      lmax[id * LSTRIDE + f]);
        }
    }
}

// The MLP is linear up to the sigmoid:
//   out[i,j,c] = sigmoid( (V@W1top@W2)[i,c] + (V@W1bot@W2)[j,c] + (b1@W2+b2)[c] )
// Stage 1 (WMMA): A = V@W1top, B = V@W1bot  (32x64 @ 64x32), f16 in / f32 acc.
// Stage 2: fold W2/b1/b2 (32-long dots). Stage 3: sigmoid + transposed store.
__global__ __launch_bounds__(256) void mlp_wmma_kernel(const float* __restrict__ vec,
                                                       const float* __restrict__ W1,
                                                       const float* __restrict__ b1,
                                                       const float* __restrict__ W2,
                                                       const float* __restrict__ b2,
                                                       float*       __restrict__ conn) {
    __shared__ float AB[2][32 * 33];   // padded stride 33
    __shared__ float S[2][32][4];
    __shared__ float Kc[4];

    const int b    = blockIdx.x;
    const int t    = threadIdx.x;
    const int lane = t & 31;
    const int wave = t >> 5;
    const int hi   = lane >> 4;        // lane group (0: lanes 0-15, 1: 16-31)
    const int lrow = lane & 15;

    const int matsel = wave >> 2;      // 0: W1 rows 0..63 (i side), 1: rows 64..127 (j side)
    const int Mt     = (wave >> 1) & 1;
    const int Nt     = wave & 1;

    const float* Vb = vec + (size_t)b * NOBJ * FDIM;

    v8f acc = {};
    #pragma unroll
    for (int ks = 0; ks < 2; ++ks) {
        v16h av, bv;
        #pragma unroll
        for (int h = 0; h < 16; ++h) {
            // A operand 16x32 f16 layout (ISA 7.12.2): K = 16*(h>=8) + 8*hi + (h&7)
            const int kA = ((h & 8) << 1) + (hi << 3) + (h & 7) + ks * 32;
            av[h] = (_Float16)Vb[(Mt * 16 + lrow) * FDIM + kA];
            // B operand 32x16 f16 layout: row K = 16*(h>=8) + 8*hi + (h&7), col N = lane%16
            const int kB = ((h & 8) << 1) + (hi << 3) + (h & 7) + ks * 32;
            bv[h] = (_Float16)W1[(matsel * 64 + kB) * 32 + (Nt * 16 + lrow)];
        }
        acc = __builtin_amdgcn_wmma_f32_16x16x32_f16(
                  /*neg_a=*/false, av, /*neg_b=*/false, bv,
                  /*c_mod=*/(short)0, acc, /*reuse_a=*/false, /*reuse_b=*/false);
    }
    #pragma unroll
    for (int v = 0; v < 8; ++v) {       // C/D layout: row = v + 8*hi, col = lane%16
        const int row = Mt * 16 + v + hi * 8;
        const int col = Nt * 16 + lrow;
        AB[matsel][row * 33 + col] = acc[v];
    }
    __syncthreads();

    {   // threads 0..127: S[i][c] from A; 128..255: T[j][c] from B
        const int side = t >> 7;
        const int i    = (t >> 2) & 31;
        const int c    = t & 3;
        float s = 0.f;
        #pragma unroll
        for (int h = 0; h < 32; ++h) s += AB[side][i * 33 + h] * W2[h * 4 + c];
        S[side][i][c] = s;
        if (t < 4) {
            float kk = b2[t];
            #pragma unroll
            for (int h = 0; h < 32; ++h) kk += b1[h] * W2[h * 4 + t];
            Kc[t] = kk;
        }
    }
    __syncthreads();

    float* connb = conn + (size_t)b * 4 * NOBJ * NOBJ;
    #pragma unroll
    for (int r = 0; r < 16; ++r) {
        const int o = t + 256 * r;          // o = ((c*32 + j)*32 + i): coalesced stores
        const int i = o & 31;
        const int j = (o >> 5) & 31;
        const int c = o >> 10;
        const float x = S[0][i][c] + S[1][j][c] + Kc[c];
        connb[o] = 1.0f / (1.0f + __expf(-x));
    }
}

extern "C" void kernel_launch(void* const* d_in, const int* in_sizes, int n_in,
                              void* d_out, int out_size, void* d_ws, size_t ws_size,
                              hipStream_t stream) {
    const float* enc   = (const float*)d_in[0];
    const int*   masks = (const int*)  d_in[1];
    const float* W1    = (const float*)d_in[2];
    const float* b1    = (const float*)d_in[3];
    const float* W2    = (const float*)d_in[4];
    const float* b2    = (const float*)d_in[5];

    float* out  = (float*)d_out;
    float* vec  = out;                          // vectors: 8*32*64 floats
    float* conn = out + BSZ * NOBJ * FDIM;      // connections: 8*4*32*32 floats

    init_vec_kernel<<<(BSZ * NOBJ * FDIM + 255) / 256, 256, 0, stream>>>(vec);
    segmax_kernel<<<BSZ * CHUNKS, 256, 0, stream>>>(enc, masks, vec);
    mlp_wmma_kernel<<<BSZ, 256, 0, stream>>>(vec, W1, b1, W2, b2, conn);
}
